// bahdanau_attention_74294344286644
// MI455X (gfx1250) — matrix operand order repaired and checked by reading, not compiled, not run on hardware
//
#include <hip/hip_runtime.h>
#include <hip/hip_bf16.h>
#include <math.h>

#define B_ 32
#define L_ 2048
#define H_ 1024
#define E_ 512

typedef __attribute__((ext_vector_type(16))) __bf16 v16bf;
typedef __attribute__((ext_vector_type(8)))  float  v8f;

__device__ __forceinline__ unsigned short f2bf(float f) {
  unsigned int u = __builtin_bit_cast(unsigned int, f);
  u += 0x7FFFu + ((u >> 16) & 1u);   // round-to-nearest-even
  return (unsigned short)(u >> 16);
}

// ---------------- kernel 0: zero scratch (scores + c_t accumulators) --------
__global__ void zero_kernel(float* __restrict__ p, int n) {
  int i = blockIdx.x * 256 + threadIdx.x;
  if (i < n) p[i] = 0.0f;
}

// ---------------- kernel 1: W_enc -> bf16 B-fragment layout -----------------
// Layout: frag[(nt*32 + kt)*32 + lane] = 16 ushorts (8 packed uints).
// B k-map (ISA 16-bit table order): half h, VGPR j packs K = (j>>2)*16 + 8h + 2(j&3),
// +1.
__global__ void wenc_pack_kernel(const float* __restrict__ W_enc,
                                 unsigned int* __restrict__ wB) {
  int nt   = blockIdx.x >> 5;    // 0..63  (N tile of 16)
  int kt   = blockIdx.x & 31;    // 0..31  (K tile of 32)
  int lane = threadIdx.x;        // 0..31
  int half = lane >> 4;
  int col  = nt * 16 + (lane & 15);
#pragma unroll
  for (int j = 0; j < 8; ++j) {
    int k0 = kt * 32 + (j >> 2) * 16 + half * 8 + 2 * (j & 3);
    float f0 = W_enc[(size_t)k0 * H_ + col];
    float f1 = W_enc[(size_t)(k0 + 1) * H_ + col];
    unsigned int pk = (unsigned int)f2bf(f0) | ((unsigned int)f2bf(f1) << 16);
    wB[(((nt * 32 + kt) * 32 + lane) * 8) + j] = pk;
  }
}

// ---------------- kernel 2: gamma_dec = h@W_dec + b_dec + b_enc -------------
__global__ void gdec_kernel(const float* __restrict__ h, const float* __restrict__ W_dec,
                            const float* __restrict__ b_dec, const float* __restrict__ b_enc,
                            float* __restrict__ gd) {
  int b = blockIdx.y;
  int k = blockIdx.x * 256 + threadIdx.x;
  float acc = b_dec[k] + b_enc[k];
  for (int j = 0; j < H_; ++j)
    acc += h[b * H_ + j] * W_dec[(size_t)j * H_ + k];
  gd[b * H_ + k] = acc;
}

// ---------------- kernel 3: fused scores (WMMA GEMM + tanh + v-dot) ---------
// Workgroup = 512 threads (16 waves), owns 32 rows of contexts for one batch.
// Each wave: 2 m-tiles x 4 n-tiles, K-loop of 32 wmma_f32_16x16x32_bf16.
__launch_bounds__(512)
__global__ void scores_kernel(const float* __restrict__ ctx,
                              const unsigned int* __restrict__ wBu,
                              const float* __restrict__ gd,
                              const float* __restrict__ W_v,
                              float* __restrict__ scores) {
  // A fragments for 2 m-tiles x 32 k-tiles: (mt*32+kt)*32+lane -> 16 ushorts
  __shared__ __align__(32) unsigned short ldsA[2 * 32 * 32 * 16]; // 64 KB
  const int b   = blockIdx.y;
  const int l0  = blockIdx.x * 32;
  const int tid = threadIdx.x;

  // ---- Phase A: contexts rows -> bf16 A-fragment layout in LDS ----
  // A k-map (ISA 16-bit A table): hi=(kk>>3)&1, j=(kk>>4)*4+((kk&7)>>1), p=kk&1
  {
    const float2* src = reinterpret_cast<const float2*>(ctx + (size_t)(b * L_ + l0) * H_);
    unsigned int* ldsU = reinterpret_cast<unsigned int*>(ldsA);
#pragma unroll
    for (int it = 0; it < 32; ++it) {
      int pid = it * 512 + tid;          // 0..16383 pairs
      int m   = pid >> 9;                // row 0..31
      int pr  = pid & 511;               // pair index in row
      float2 v = src[m * 512 + pr];
      int k  = pr * 2;
      int kt = k >> 5;
      int kk = k & 31;
      int hi = (kk >> 3) & 1;
      int j  = (kk >> 4) * 4 + ((kk & 7) >> 1);
      int mt = m >> 4;
      int lane = (hi << 4) | (m & 15);
      unsigned int pk = (unsigned int)f2bf(v.x) | ((unsigned int)f2bf(v.y) << 16);
      ldsU[(((mt * 32 + kt) * 32 + lane) * 8) + j] = pk;
    }
  }
  __syncthreads();

  // ---- Phase B: WMMA ----
  const int wv_  = tid >> 5;     // wave 0..15
  const int lane = tid & 31;
  const v16bf* aF = reinterpret_cast<const v16bf*>(ldsA);
  const v16bf* bF = reinterpret_cast<const v16bf*>(wBu);

  v8f acc0[4], acc1[4];
#pragma unroll
  for (int cc = 0; cc < 4; ++cc) {
    acc0[cc] = (v8f){0.f,0.f,0.f,0.f,0.f,0.f,0.f,0.f};
    acc1[cc] = (v8f){0.f,0.f,0.f,0.f,0.f,0.f,0.f,0.f};
  }

  for (int kt = 0; kt < 32; ++kt) {
    v16bf a0 = aF[(0  + kt) * 32 + lane];
    v16bf a1 = aF[(32 + kt) * 32 + lane];
#pragma unroll
    for (int cc = 0; cc < 4; ++cc) {
      int nt = wv_ * 4 + cc;
      v16bf bf = bF[((nt * 32 + kt) * 32) + lane];
      acc0[cc] = __builtin_amdgcn_wmma_f32_16x16x32_bf16(
          false, a0, false, bf, (short)0, acc0[cc], false, false);
      acc1[cc] = __builtin_amdgcn_wmma_f32_16x16x32_bf16(
          false, a1, false, bf, (short)0, acc1[cc], false, false);
    }
  }

  // ---- Epilogue: tanh(acc + gd) * W_v, reduce over H ----
  const int half = lane >> 4;
  float rs0[8] = {0,0,0,0,0,0,0,0};
  float rs1[8] = {0,0,0,0,0,0,0,0};
#pragma unroll
  for (int cc = 0; cc < 4; ++cc) {
    int col   = (wv_ * 4 + cc) * 16 + (lane & 15);
    float gdv = gd[b * H_ + col];
    float wvv = W_v[col];
#pragma unroll
    for (int r = 0; r < 8; ++r) {
      rs0[r] += tanhf(acc0[cc][r] + gdv) * wvv;   // row = 8*half + r (m-tile 0)
      rs1[r] += tanhf(acc1[cc][r] + gdv) * wvv;   // row = 8*half + r (m-tile 1)
    }
  }
#pragma unroll
  for (int r = 0; r < 8; ++r) {
    float v0 = rs0[r], v1 = rs1[r];
    v0 += __shfl_xor(v0, 1); v0 += __shfl_xor(v0, 2);
    v0 += __shfl_xor(v0, 4); v0 += __shfl_xor(v0, 8);
    v1 += __shfl_xor(v1, 1); v1 += __shfl_xor(v1, 2);
    v1 += __shfl_xor(v1, 4); v1 += __shfl_xor(v1, 8);
    if ((lane & 15) == 0) {
      int row = half * 8 + r;
      atomicAdd(&scores[(size_t)b * L_ + l0 + row],      v0);
      atomicAdd(&scores[(size_t)b * L_ + l0 + 16 + row], v1);
    }
  }
}

// ---------------- kernel 4: softmax over L per batch ------------------------
__global__ void softmax_kernel(const float* __restrict__ scores,
                               const float* __restrict__ b_v,
                               float* __restrict__ weights) {
  __shared__ float red[256];
  int b = blockIdx.x, t = threadIdx.x;
  float bv = b_v[0];
  float vals[8];
  float mx = -INFINITY;
#pragma unroll
  for (int i = 0; i < 8; ++i) {
    vals[i] = scores[b * L_ + i * 256 + t] + bv;
    mx = fmaxf(mx, vals[i]);
  }
  red[t] = mx; __syncthreads();
  for (int s = 128; s > 0; s >>= 1) { if (t < s) red[t] = fmaxf(red[t], red[t + s]); __syncthreads(); }
  mx = red[0]; __syncthreads();
  float sm = 0.f;
#pragma unroll
  for (int i = 0; i < 8; ++i) { vals[i] = __expf(vals[i] - mx); sm += vals[i]; }
  red[t] = sm; __syncthreads();
  for (int s = 128; s > 0; s >>= 1) { if (t < s) red[t] += red[t + s]; __syncthreads(); }
  float inv = 1.0f / red[0];
#pragma unroll
  for (int i = 0; i < 8; ++i) weights[b * L_ + i * 256 + t] = vals[i] * inv;
}

// ---------------- kernel 5: c_t = sum_l weights * contexts ------------------
__global__ void ct_kernel(const float* __restrict__ weights,
                          const float* __restrict__ ctx,
                          float* __restrict__ ct) {
  int hblk = blockIdx.x, lch = blockIdx.y, b = blockIdx.z;
  int hcol = hblk * 256 + threadIdx.x;
  const float* cb = ctx + ((size_t)(b * L_ + lch * 256)) * H_ + hcol;
  const float* wb = weights + b * L_ + lch * 256;
  float acc = 0.f;
  for (int l = 0; l < 256; ++l) acc += wb[l] * cb[(size_t)l * H_];
  atomicAdd(&ct[b * H_ + hcol], acc);
}

// ---------------- kernel 6: r_t = [c_t,h,x]@W_r + b_r, then maxout ----------
__global__ void out_kernel(const float* __restrict__ ct, const float* __restrict__ h,
                           const float* __restrict__ x, const float* __restrict__ W_r,
                           const float* __restrict__ b_r, float* __restrict__ out) {
  int b = blockIdx.y;
  int n = blockIdx.x * 256 + threadIdx.x;     // 0..2047
  float acc = b_r[n];
  const float* wr = W_r + n;
  for (int k = 0; k < H_; ++k) acc += ct[b * H_ + k] * wr[(size_t)k * (2 * H_)];
  for (int k = 0; k < H_; ++k) acc += h[b * H_ + k]  * wr[(size_t)(H_ + k) * (2 * H_)];
  for (int k = 0; k < E_; ++k) acc += x[b * E_ + k]  * wr[(size_t)(2 * H_ + k) * (2 * H_)];
  float other = __shfl_xor(acc, 1);
  if ((n & 1) == 0) out[b * H_ + (n >> 1)] = fmaxf(acc, other);
}

// ---------------- launch ----------------------------------------------------
extern "C" void kernel_launch(void* const* d_in, const int* in_sizes, int n_in,
                              void* d_out, int out_size, void* d_ws, size_t ws_size,
                              hipStream_t stream) {
  const float* h_      = (const float*)d_in[0];
  const float* x_      = (const float*)d_in[1];
  const float* ctx     = (const float*)d_in[2];
  const float* W_enc   = (const float*)d_in[3];
  const float* b_enc   = (const float*)d_in[4];
  const float* W_dec   = (const float*)d_in[5];
  const float* b_dec   = (const float*)d_in[6];
  const float* W_v     = (const float*)d_in[7];
  const float* b_v     = (const float*)d_in[8];
  const float* W_r     = (const float*)d_in[9];
  const float* b_r     = (const float*)d_in[10];

  float* out = (float*)d_out;                  // [B,H] output then [B,L] weights
  float* weights = out + B_ * H_;

  char* ws = (char*)d_ws;
  const size_t WENC_OFF   = 0;                                   // 2 MB bf16 fragments
  const size_t GD_OFF     = WENC_OFF + (size_t)64*32*32*16*2;    // [B,H] fp32
  const size_t SCORES_OFF = GD_OFF + (size_t)B_*H_*4;            // [B,L] fp32
  const size_t CT_OFF     = SCORES_OFF + (size_t)B_*L_*4;        // [B,H] fp32
  unsigned int* wB   = (unsigned int*)(ws + WENC_OFF);
  float* gd     = (float*)(ws + GD_OFF);
  float* scores = (float*)(ws + SCORES_OFF);
  float* ct     = (float*)(ws + CT_OFF);

  // 0) zero scores + c_t accumulators (contiguous)
  zero_kernel<<<dim3((B_*L_ + B_*H_ + 255) / 256), 256, 0, stream>>>(scores, B_*L_ + B_*H_);
  // 1) pack W_enc to bf16 WMMA-B fragment layout
  wenc_pack_kernel<<<dim3(64 * 32), 32, 0, stream>>>(W_enc, wB);
  // 2) gamma_dec (+ both biases)
  gdec_kernel<<<dim3(H_ / 256, B_), 256, 0, stream>>>(h_, W_dec, b_dec, b_enc, gd);
  // 3) fused WMMA scores
  scores_kernel<<<dim3(L_ / 32, B_), 512, 0, stream>>>(ctx, wB, gd, W_v, scores);
  // 4) softmax -> weights (second output)
  softmax_kernel<<<dim3(B_), 256, 0, stream>>>(scores, b_v, weights);
  // 5) context vector
  ct_kernel<<<dim3(H_ / 256, 8, B_), 256, 0, stream>>>(weights, ctx, ct);
  // 6) final GEMM + maxout -> first output
  out_kernel<<<dim3((2 * H_) / 256, B_), 256, 0, stream>>>(ct, h_, x_, W_r, b_r, out);
}
